// DPMACE_68247030333589
// MI455X (gfx1250) — matrix operand order repaired and checked by reading, not compiled, not run on hardware
//
#include <hip/hip_runtime.h>
#include <hip/hip_bf16.h>

// ---------------- problem constants ----------------
#define NN 20000      // nodes
#define NE 320000     // edges
#define CC 32         // channels
#define SS 9          // spherical harmonics
#define ZZ 10         // species
#define GG 16         // graphs
#define NBR 8         // radial bases
#define HH 64         // MLP hidden
#define CS 288        // C*S
#define RMAXF 5.0f
#define PI_F 3.14159265358979f

// swizzled weight image offsets (ushort units) within one layer's block
#define OFF_W1P 0        // 32x64  B-frags  (K padded 8->32)
#define OFF_W2F 2048     // 64x64  B-frags
#define OFF_W3F 6144     // 64x288 B-frags
#define OFF_W3T 24576    // 288x64 B-frags (transposed)
#define OFF_W2T 43008    // 64x64  B-frags (transposed)
#define OFF_W1T 47104    // 64x16  B-frags (transposed, N padded 8->16)
#define OFF_WLP 48128    // 288x32 B-frags
#define OFF_WLPT 57344   // 32x288 B-frags (transposed)
#define OFF_WUP 66560    // 32x32  B-frags
#define LAYER_USH 67584  // per-layer stride in ushorts

typedef __attribute__((ext_vector_type(16))) __bf16 v16bf;
typedef __attribute__((ext_vector_type(8)))  float  v8f;

union V16U { v16bf v; unsigned short u[16]; };
static_assert(sizeof(v16bf) == 32, "v16bf size");

__device__ __forceinline__ unsigned short f2bf(float f) {
    union { float f; unsigned u; } x; x.f = f;
    unsigned r = x.u + 0x7fffu + ((x.u >> 16) & 1u);   // round-to-nearest-even
    return (unsigned short)(r >> 16);
}
__device__ __forceinline__ float siluf(float x)  { float s = 1.f/(1.f+__expf(-x)); return x*s; }
__device__ __forceinline__ float dsiluf(float x) { float s = 1.f/(1.f+__expf(-x)); return s + x*s*(1.f-s); }
__device__ __forceinline__ v8f vzero() { v8f z = {0.f,0.f,0.f,0.f,0.f,0.f,0.f,0.f}; return z; }

__device__ __forceinline__ v8f wmma_bf16(v16bf a, v16bf b, v8f c) {
    return __builtin_amdgcn_wmma_f32_16x16x32_bf16(false, a, false, b, (short)0, c, false, false);
}

// fragment-major index for B layout: element (k,n) of a (K x N) matrix, ntiles = N/16.
// In-fragment: lane = 16*(k_local>>4) + (n&15), slot j = k_local&15.
__device__ __forceinline__ int b_idx(int k, int n, int ntiles) {
    return (((k >> 5) * ntiles + (n >> 4)) << 9) + (((((k >> 4) & 1) << 4) + (n & 15)) << 4) + (k & 15);
}
// A-fragment index within one 16x32 chunk: element (row, k).
// lane = row + 16*((k>>4)&1), slot j = k&15.
__device__ __forceinline__ int a_idx(int row, int k) {
    return (((((k >> 4) & 1) << 4) + row) << 4) + (k & 15);
}
// contiguous 32-byte fragment load (LDS or global): two b128 ops
__device__ __forceinline__ v16bf ld_frag(const unsigned short* buf, int frag, int lane) {
    return *(const v16bf*)(buf + (frag << 9) + (lane << 4));
}
// wide block copy (global->LDS), cnt in ushorts (multiple of 8)
__device__ __forceinline__ void cp2lds(unsigned short* dst, const unsigned short* __restrict__ src,
                                       int cnt, int tid, int nthr) {
    uint4* d = (uint4*)dst; const uint4* s = (const uint4*)src;
    int n = cnt >> 3;
    for (int i = tid; i < n; i += nthr) d[i] = s[i];
}

// ---------------- weight swizzler: f32 (opt transposed, zero-padded) -> B-fragment bf16 -------
__global__ void k_swz(const float* __restrict__ src, unsigned short* __restrict__ dst,
                      int srcRows, int srcCols, int outK, int outN, int transp) {
    int idx = blockIdx.x * blockDim.x + threadIdx.x;
    if (idx >= outK * outN) return;
    int k = idx / outN, n = idx - k * outN;
    float v = 0.f;
    if (transp) { if (n < srcRows && k < srcCols) v = src[n * srcCols + k]; }
    else        { if (k < srcRows && n < srcCols) v = src[k * srcCols + n]; }
    dst[b_idx(k, n, outN >> 4)] = f2bf(v);
}

// ---------------- forward: node init ----------------
__global__ void k_node_init(const float* __restrict__ attrs, const float* __restrict__ ae,
                            const float* __restrict__ Wemb, const int* __restrict__ batch,
                            float* h0, float* ne0, int* spec, float* e0g) {
    int n = blockIdx.x * blockDim.x + threadIdx.x;
    if (n >= NN) return;
    int sp = 0; float best = -1.f, e0 = 0.f;
    for (int z = 0; z < ZZ; ++z) {
        float a = attrs[n * ZZ + z];
        e0 += a * ae[z];
        if (a > best) { best = a; sp = z; }
    }
    spec[n] = sp; ne0[n] = e0;
    for (int c = 0; c < CC; ++c) {
        float s = 0.f;
        for (int z = 0; z < ZZ; ++z) s += attrs[n * ZZ + z] * Wemb[z * CC + c];
        h0[n * CC + c] = s;
    }
    atomicAdd(&e0g[batch[n]], e0);
}

// ---------------- forward: edge geometry ----------------
__global__ void k_edge_geom(const float* __restrict__ pos, const float* __restrict__ shifts,
                            const int* __restrict__ eidx, float* u, float* invr, float* Y, float* Rr) {
    int e = blockIdx.x * blockDim.x + threadIdx.x;
    if (e >= NE) return;
    int sd = eidx[e], rc = eidx[NE + e];
    float vx = pos[rc*3+0] - pos[sd*3+0] + shifts[e*3+0];
    float vy = pos[rc*3+1] - pos[sd*3+1] + shifts[e*3+1];
    float vz = pos[rc*3+2] - pos[sd*3+2] + shifts[e*3+2];
    float r = sqrtf(vx*vx + vy*vy + vz*vz + 1e-12f);
    float ir = 1.f / r;
    float x = vx*ir, y = vy*ir, z = vz*ir;
    u[e*3+0] = x; u[e*3+1] = y; u[e*3+2] = z; invr[e] = ir;
    const float c1 = 1.73205081f, c2 = 3.87298335f, c3 = 1.11803399f;
    Y[e*SS+0] = 1.f;                 Y[e*SS+1] = c1*x;   Y[e*SS+2] = c1*y;
    Y[e*SS+3] = c1*z;                Y[e*SS+4] = c2*x*y; Y[e*SS+5] = c2*y*z;
    Y[e*SS+6] = c3*(3.f*z*z - 1.f);  Y[e*SS+7] = c2*x*z; Y[e*SS+8] = 0.5f*c2*(x*x - y*y);
    float t = r * (1.f / RMAXF), env = 0.f;
    if (t < 1.f) {
        float t2 = t*t, t4 = t2*t2, t6 = t4*t2, t7 = t6*t, t8 = t7*t;
        env = 1.f - 28.f*t6 + 48.f*t7 - 21.f*t8;
    }
    const float Ac = 0.632455532f;   // sqrt(2/R_MAX)
    for (int k = 0; k < NBR; ++k) {
        float a = (float)(k+1) * (PI_F / RMAXF);
        Rr[e*NBR + k] = Ac * __sinf(a * r) * ir * env;
    }
}

// ---------------- fold W_lin @ W_prod ----------------
__global__ void k_wlp(const float* __restrict__ Wlin, const float* __restrict__ Wprod, float* Wlp) {
    int idx = blockIdx.x * blockDim.x + threadIdx.x;
    if (idx >= 2 * CS * CC) return;
    int l = idx / (CS * CC), rem = idx - l * CS * CC, i = rem / CC, d = rem % CC;
    float s = 0.f;
    for (int c = 0; c < CC; ++c)
        s += Wlin[l*CS*CC + i*CC + c] * Wprod[l*CC*CC + c*CC + d];
    Wlp[idx] = s;
}

// ---------------- hu = h @ W_up (WMMA) ----------------
__global__ void __launch_bounds__(256) k_hu(const float* __restrict__ h,
                                            const unsigned short* __restrict__ Wsw, float* hu) {
    __shared__ alignas(32) unsigned short Wl[1024];
    __shared__ alignas(32) unsigned short Hst[8][512];
    int tid = threadIdx.x, lane = tid & 31, wv = tid >> 5;
    cp2lds(Wl, Wsw + OFF_WUP, 1024, tid, 256);
    __syncthreads();
    int tile = blockIdx.x * 8 + wv;
    if (tile * 16 >= NN) return;
    int r0 = tile * 16, col = lane & 15, half = lane >> 4;
    unsigned short* hs = &Hst[wv][0];
    for (int i = lane; i < 512; i += 32) { int m = i >> 5, c = i & 31; hs[a_idx(m, c)] = f2bf(h[(r0+m)*CC + c]); }
    v16bf A = ld_frag(hs, 0, lane);
    for (int nt = 0; nt < 2; ++nt) {
        v8f acc = wmma_bf16(A, ld_frag(Wl, nt, lane), vzero());
        for (int v = 0; v < 8; ++v) { int m = v + 8*half; hu[(r0+m)*CC + nt*16 + col] = acc[v]; }
    }
}

// ---------------- fused edge forward: R->64->64->288 MLP, *hu[sender]*Y, scatter ----------
__global__ void __launch_bounds__(64) k_edge_msg(const unsigned short* __restrict__ Wsw,
                                                 const float* __restrict__ hu, const float* __restrict__ Yg,
                                                 const float* __restrict__ Rg, const int* __restrict__ eidx,
                                                 float* msg) {
    __shared__ alignas(32) unsigned short W1P[2048];
    __shared__ alignas(32) unsigned short W2F[4096];
    __shared__ alignas(32) unsigned short W3F[18432];
    __shared__ alignas(32) unsigned short Ast[2][512];
    __shared__ alignas(32) unsigned short Tst[2][1024];
    __shared__ float Yl[2][144], HUS[2][512];
    __shared__ int RID[2][16], SID[2][16];
    int tid = threadIdx.x, lane = tid & 31, wv = tid >> 5;
    cp2lds(W1P, Wsw + OFF_W1P, 2048, tid, 64);
    cp2lds(W2F, Wsw + OFF_W2F, 4096, tid, 64);
    cp2lds(W3F, Wsw + OFF_W3F, 18432, tid, 64);
    __syncthreads();
    int gw = blockIdx.x * 2 + wv, nw = gridDim.x * 2, ntile = NE / 16;
    unsigned short* ast = &Ast[wv][0]; unsigned short* tst = &Tst[wv][0];
    float* yl = &Yl[wv][0]; float* hus = &HUS[wv][0];
    int* rid = &RID[wv][0]; int* sid = &SID[wv][0];
    int col = lane & 15, half = lane >> 4;
    for (int t = gw; t < ntile; t += nw) {
        int e0 = t * 16;
        if (lane < 16) { rid[lane] = eidx[NE + e0 + lane]; sid[lane] = eidx[e0 + lane]; }
        for (int i = lane; i < 512; i += 32) { int m = i >> 5, k = i & 31; ast[a_idx(m, k)] = (k < NBR) ? f2bf(Rg[(e0+m)*NBR + k]) : (unsigned short)0; }
        for (int i = lane; i < 144; i += 32) { int m = i / SS, s = i % SS; yl[i] = Yg[(e0+m)*SS + s]; }
        for (int i = lane; i < 128; i += 32) { int m = i >> 3, q = i & 7; ((float4*)hus)[i] = ((const float4*)(hu + (size_t)sid[m]*CC))[q]; }
        // t1 = silu(R @ W1)
        v16bf A = ld_frag(ast, 0, lane);
        for (int nt = 0; nt < 4; ++nt) {
            v8f a = wmma_bf16(A, ld_frag(W1P, nt, lane), vzero());
            for (int v = 0; v < 8; ++v) { int m = v + 8*half; tst[((nt>>1)<<9) + a_idx(m, (nt&1)*16 + col)] = f2bf(siluf(a[v])); }
        }
        // t2 = silu(t1 @ W2)
        v16bf A2a = ld_frag(tst, 0, lane), A2b = ld_frag(tst, 1, lane);
        for (int nt = 0; nt < 4; ++nt) {
            v8f a = wmma_bf16(A2a, ld_frag(W2F, nt, lane), vzero());
            a = wmma_bf16(A2b, ld_frag(W2F, 4 + nt, lane), a);
            for (int v = 0; v < 8; ++v) { int m = v + 8*half; tst[((nt>>1)<<9) + a_idx(m, (nt&1)*16 + col)] = f2bf(siluf(a[v])); }
        }
        // tpw tile-by-tile, fused elementwise + atomic scatter
        v16bf A3a = ld_frag(tst, 0, lane), A3b = ld_frag(tst, 1, lane);
        for (int nt = 0; nt < 18; ++nt) {
            v8f a = wmma_bf16(A3a, ld_frag(W3F, nt, lane), vzero());
            a = wmma_bf16(A3b, ld_frag(W3F, 18 + nt, lane), a);
            int n = nt*16 + col, c = n / SS, s = n - SS*c;
            for (int v = 0; v < 8; ++v) {
                int m = v + 8*half;
                float mji = hus[m*CC + c] * yl[m*SS + s] * a[v];
                atomicAdd(&msg[(size_t)rid[m]*CS + n], mji);
            }
        }
    }
}

// ---------------- skip connection ----------------
__global__ void k_sc(const float* __restrict__ h, const float* __restrict__ Wskip,
                     const int* __restrict__ spec, float* sc) {
    int idx = blockIdx.x * blockDim.x + threadIdx.x;
    if (idx >= NN * CC) return;
    int n = idx >> 5, d = idx & 31;
    const float* W = Wskip + (size_t)spec[n] * CC * CC;
    float s = 0.f;
    for (int c = 0; c < CC; ++c) s += h[n*CC + c] * W[c*CC + d];
    sc[idx] = s;
}

// ---------------- h_next = (msg @ Wlp)/16 + sc (WMMA, K=288) ----------------
__global__ void __launch_bounds__(256) k_node_update(const float* __restrict__ msg,
                                                     const unsigned short* __restrict__ Wsw,
                                                     const float* __restrict__ sc, float* hout, float* feats) {
    __shared__ alignas(32) unsigned short Wl[CS * CC];
    __shared__ alignas(32) unsigned short Mst[8][512];
    int tid = threadIdx.x, lane = tid & 31, wv = tid >> 5;
    cp2lds(Wl, Wsw + OFF_WLP, CS * CC, tid, 256);
    __syncthreads();
    int tile = blockIdx.x * 8 + wv;
    if (tile * 16 >= NN) return;
    int r0 = tile * 16, col = lane & 15, half = lane >> 4;
    unsigned short* mst = &Mst[wv][0];
    v8f acc[2] = { vzero(), vzero() };
    for (int kc = 0; kc < 9; ++kc) {
        for (int i = lane; i < 512; i += 32) { int m = i >> 5, k = i & 31; mst[a_idx(m, k)] = f2bf(msg[(size_t)(r0+m)*CS + kc*32 + k]); }
        v16bf A = ld_frag(mst, 0, lane);
        for (int nt = 0; nt < 2; ++nt) acc[nt] = wmma_bf16(A, ld_frag(Wl, kc*2 + nt, lane), acc[nt]);
    }
    for (int nt = 0; nt < 2; ++nt)
        for (int v = 0; v < 8; ++v) {
            int m = v + 8*half, n = nt*16 + col, row = r0 + m;
            float val = acc[nt][v] * (1.f/16.f) + sc[row*CC + n];
            hout[row*CC + n] = val;
            feats[(size_t)row*64 + n] = val;   // feats base pre-offset by layer*32
        }
}

// ---------------- es += h @ W_read ----------------
__global__ void k_es(const float* __restrict__ h, const float* __restrict__ Wread, float* es) {
    int n = blockIdx.x * blockDim.x + threadIdx.x;
    if (n >= NN) return;
    float s = 0.f;
    for (int c = 0; c < CC; ++c) s += h[n*CC + c] * Wread[c];
    es[n] += s;
}

__global__ void k_readout(const float* __restrict__ es, const float* __restrict__ ne0,
                          const float* __restrict__ scale, const float* __restrict__ shiftv,
                          const int* __restrict__ batch, float* node_energy, float* ieg) {
    int n = blockIdx.x * blockDim.x + threadIdx.x;
    if (n >= NN) return;
    float nies = scale[0] * es[n] + shiftv[0];
    node_energy[n] = ne0[n] + nies;
    atomicAdd(&ieg[batch[n]], nies);
}
__global__ void k_final(const float* e0g, const float* ieg, float* te, float* ie) {
    int g = threadIdx.x;
    if (g < GG) { te[g] = e0g[g] + ieg[g]; ie[g] = ieg[g]; }
}

// ---------------- backward ----------------
__global__ void k_ginit(const float* __restrict__ Wread1, const float* __restrict__ scale, float* gh) {
    int idx = blockIdx.x * blockDim.x + threadIdx.x;
    if (idx >= NN * CC) return;
    gh[idx] = scale[0] * Wread1[idx & 31];
}

// g_msg = (g_h @ Wlp^T)/16  (WMMA on pre-transposed swizzled image)
__global__ void __launch_bounds__(256) k_gmsg(const float* __restrict__ gh,
                                              const unsigned short* __restrict__ Wsw, float* gmsg) {
    __shared__ alignas(32) unsigned short Wl[CS * CC];
    __shared__ alignas(32) unsigned short Gst[8][512];
    int tid = threadIdx.x, lane = tid & 31, wv = tid >> 5;
    cp2lds(Wl, Wsw + OFF_WLPT, CS * CC, tid, 256);
    __syncthreads();
    int tile = blockIdx.x * 8 + wv;
    if (tile * 16 >= NN) return;
    int r0 = tile * 16, col = lane & 15, half = lane >> 4;
    unsigned short* gs = &Gst[wv][0];
    for (int i = lane; i < 512; i += 32) { int m = i >> 5, c = i & 31; gs[a_idx(m, c)] = f2bf(gh[(r0+m)*CC + c]); }
    v16bf A = ld_frag(gs, 0, lane);
    for (int nt = 0; nt < 18; ++nt) {
        v8f acc = wmma_bf16(A, ld_frag(Wl, nt, lane), vzero());
        for (int v = 0; v < 8; ++v) { int m = v + 8*half; gmsg[(size_t)(r0+m)*CS + nt*16 + col] = acc[v] * (1.f/16.f); }
    }
}

// backward stage 1: recompute t1/t2/tpw, emit g_tpw fragments to HBM, g_hu + angular forces
__global__ void __launch_bounds__(32) k_edge_b1(const unsigned short* __restrict__ Wsw,
                                                const float* __restrict__ hu, const float* __restrict__ Yg,
                                                const float* __restrict__ Rg, const float* __restrict__ Ug,
                                                const float* __restrict__ invrG, const int* __restrict__ eidx,
                                                const float* __restrict__ gmsg, unsigned short* __restrict__ gtpw,
                                                float* forces, float* ghu, int do_ghu) {
    __shared__ alignas(32) unsigned short W1P[2048];
    __shared__ alignas(32) unsigned short W2F[4096];
    __shared__ alignas(32) unsigned short W3F[18432];
    __shared__ alignas(32) unsigned short Ast[512];
    __shared__ alignas(32) unsigned short Tst[1024];
    __shared__ alignas(32) unsigned short Gst[512];
    __shared__ float Yl[144], HUS[512], GHUl[512], GYl[144], Ul[48], IVl[16];
    __shared__ int RID[16], SID[16];
    int lane = threadIdx.x, col = lane & 15, half = lane >> 4;
    cp2lds(W1P, Wsw + OFF_W1P, 2048, lane, 32);
    cp2lds(W2F, Wsw + OFF_W2F, 4096, lane, 32);
    cp2lds(W3F, Wsw + OFF_W3F, 18432, lane, 32);
    int ntile = NE / 16, nw = gridDim.x;
    for (int t = blockIdx.x; t < ntile; t += nw) {
        int e0 = t * 16;
        if (lane < 16) {
            RID[lane] = eidx[NE + e0 + lane]; SID[lane] = eidx[e0 + lane];
            IVl[lane] = invrG[e0 + lane];
            Ul[lane*3+0] = Ug[(e0+lane)*3+0]; Ul[lane*3+1] = Ug[(e0+lane)*3+1]; Ul[lane*3+2] = Ug[(e0+lane)*3+2];
        }
        for (int i = lane; i < 512; i += 32) { int m = i >> 5, k = i & 31; Ast[a_idx(m, k)] = (k < NBR) ? f2bf(Rg[(e0+m)*NBR + k]) : (unsigned short)0; GHUl[i] = 0.f; }
        for (int i = lane; i < 144; i += 32) { int m = i / SS, s = i % SS; Yl[i] = Yg[(e0+m)*SS + s]; GYl[i] = 0.f; }
        for (int i = lane; i < 128; i += 32) { int m = i >> 3, q = i & 7; ((float4*)HUS)[i] = ((const float4*)(hu + (size_t)SID[m]*CC))[q]; }
        // recompute t1, t2 (values only)
        v16bf A = ld_frag(Ast, 0, lane);
        for (int nt = 0; nt < 4; ++nt) {
            v8f a = wmma_bf16(A, ld_frag(W1P, nt, lane), vzero());
            for (int v = 0; v < 8; ++v) { int m = v + 8*half; Tst[((nt>>1)<<9) + a_idx(m, (nt&1)*16 + col)] = f2bf(siluf(a[v])); }
        }
        v16bf A2a = ld_frag(Tst, 0, lane), A2b = ld_frag(Tst, 1, lane);
        for (int nt = 0; nt < 4; ++nt) {
            v8f a = wmma_bf16(A2a, ld_frag(W2F, nt, lane), vzero());
            a = wmma_bf16(A2b, ld_frag(W2F, 4 + nt, lane), a);
            for (int v = 0; v < 8; ++v) { int m = v + 8*half; Tst[((nt>>1)<<9) + a_idx(m, (nt&1)*16 + col)] = f2bf(siluf(a[v])); }
        }
        v16bf A3a = ld_frag(Tst, 0, lane), A3b = ld_frag(Tst, 1, lane);
        // per 32-wide K chunk: tpw tiles -> g_tpw fragment to HBM, g_hu/g_Y LDS reductions
        unsigned short* gout = gtpw + (size_t)t * 4608;
        for (int pt = 0; pt < 9; ++pt) {
            for (int sub = 0; sub < 2; ++sub) {
                int nt = pt*2 + sub;
                v8f a = wmma_bf16(A3a, ld_frag(W3F, nt, lane), vzero());
                a = wmma_bf16(A3b, ld_frag(W3F, 18 + nt, lane), a);
                int n = nt*16 + col, c = n / SS, s = n - SS*c;
                for (int v = 0; v < 8; ++v) {
                    int m = v + 8*half;
                    float gm = gmsg[(size_t)RID[m]*CS + n];
                    float huv = HUS[m*CC + c], yv = Yl[m*SS + s], tp = a[v];
                    atomicAdd(&GHUl[m*CC + c], gm * yv * tp);    // ds_add_f32
                    atomicAdd(&GYl[m*SS + s], gm * huv * tp);
                    Gst[a_idx(m, sub*16 + col)] = f2bf(gm * huv * yv);
                }
            }
            // dump fragment chunk (contiguous 32B per lane)
            uint4* d = (uint4*)(gout + pt*512 + lane*16);
            const uint4* sfr = (const uint4*)(Gst + lane*16);
            d[0] = sfr[0]; d[1] = sfr[1];
        }
        // angular part of forces from g_Y
        if (lane < 16) {
            int m = lane;
            float ir = IVl[m];
            float ux = Ul[m*3], uy = Ul[m*3+1], uz = Ul[m*3+2];
            const float c1 = 1.73205081f, c2 = 3.87298335f, c3 = 1.11803399f;
            float g1=GYl[m*SS+1], g2=GYl[m*SS+2], g3=GYl[m*SS+3], g4=GYl[m*SS+4];
            float g5=GYl[m*SS+5], g6=GYl[m*SS+6], g7=GYl[m*SS+7], g8=GYl[m*SS+8];
            float gux = c1*g1 + c2*uy*g4 + c2*uz*g7 + c2*ux*g8;
            float guy = c1*g2 + c2*ux*g4 + c2*uz*g5 - c2*uy*g8;
            float guz = c1*g3 + c2*uy*g5 + 6.f*c3*uz*g6 + c2*ux*g7;
            float dg = ux*gux + uy*guy + uz*guz;
            float gvx = ir*(gux - ux*dg), gvy = ir*(guy - uy*dg), gvz = ir*(guz - uz*dg);
            int rc = RID[m], sd = SID[m];
            atomicAdd(&forces[rc*3+0], -gvx); atomicAdd(&forces[sd*3+0], gvx);
            atomicAdd(&forces[rc*3+1], -gvy); atomicAdd(&forces[sd*3+1], gvy);
            atomicAdd(&forces[rc*3+2], -gvz); atomicAdd(&forces[sd*3+2], gvz);
        }
        if (do_ghu)
            for (int i = lane; i < 512; i += 32) { int m = i >> 5, c = i & 31; atomicAdd(&ghu[(size_t)SID[m]*CC + c], GHUl[i]); }
    }
}

// backward stage 2: transposed GEMM chain g_tpw -> g_t2 -> g_t1 -> g_R, radial forces
__global__ void __launch_bounds__(32) k_edge_b2(const unsigned short* __restrict__ Wsw,
                                                const float* __restrict__ Rg, const float* __restrict__ Ug,
                                                const float* __restrict__ invrG, const int* __restrict__ eidx,
                                                const unsigned short* __restrict__ gtpw, float* forces) {
    __shared__ alignas(32) unsigned short W1P[2048];
    __shared__ alignas(32) unsigned short W2F[4096];
    __shared__ alignas(32) unsigned short W3T[18432];
    __shared__ alignas(32) unsigned short W2T[4096];
    __shared__ alignas(32) unsigned short W1T[1024];
    __shared__ alignas(32) unsigned short Ast[512];
    __shared__ alignas(32) unsigned short Tst[1024];
    __shared__ alignas(32) unsigned short G2st[1024];
    __shared__ float GRl[128], Ul[48], IVl[16];
    __shared__ int RID[16], SID[16];
    int lane = threadIdx.x, col = lane & 15, half = lane >> 4;
    cp2lds(W1P, Wsw + OFF_W1P, 2048, lane, 32);
    cp2lds(W2F, Wsw + OFF_W2F, 4096, lane, 32);
    cp2lds(W3T, Wsw + OFF_W3T, 18432, lane, 32);
    cp2lds(W2T, Wsw + OFF_W2T, 4096, lane, 32);
    cp2lds(W1T, Wsw + OFF_W1T, 1024, lane, 32);
    int ntile = NE / 16, nw = gridDim.x;
    for (int t = blockIdx.x; t < ntile; t += nw) {
        int e0 = t * 16;
        if (lane < 16) {
            RID[lane] = eidx[NE + e0 + lane]; SID[lane] = eidx[e0 + lane];
            IVl[lane] = invrG[e0 + lane];
            Ul[lane*3+0] = Ug[(e0+lane)*3+0]; Ul[lane*3+1] = Ug[(e0+lane)*3+1]; Ul[lane*3+2] = Ug[(e0+lane)*3+2];
        }
        for (int i = lane; i < 512; i += 32) { int m = i >> 5, k = i & 31; Ast[a_idx(m, k)] = (k < NBR) ? f2bf(Rg[(e0+m)*NBR + k]) : (unsigned short)0; }
        // recompute pre-activations z1, z2 (registers, C/D layout)
        v16bf A = ld_frag(Ast, 0, lane);
        v8f z1r[4];
        for (int nt = 0; nt < 4; ++nt) {
            v8f a = wmma_bf16(A, ld_frag(W1P, nt, lane), vzero());
            z1r[nt] = a;
            for (int v = 0; v < 8; ++v) { int m = v + 8*half; Tst[((nt>>1)<<9) + a_idx(m, (nt&1)*16 + col)] = f2bf(siluf(a[v])); }
        }
        v16bf A2a = ld_frag(Tst, 0, lane), A2b = ld_frag(Tst, 1, lane);
        v8f z2r[4];
        for (int nt = 0; nt < 4; ++nt) {
            v8f a = wmma_bf16(A2a, ld_frag(W2F, nt, lane), vzero());
            z2r[nt] = wmma_bf16(A2b, ld_frag(W2F, 4 + nt, lane), a);
        }
        // g_t2 = (g_tpw @ W3^T): stream g_tpw fragments straight from HBM
        const unsigned short* gin = gtpw + (size_t)t * 4608;
        v8f gt2[4] = { vzero(), vzero(), vzero(), vzero() };
        for (int kc = 0; kc < 9; ++kc) {
            v16bf Ag = ld_frag(gin, kc, lane);
            for (int nt = 0; nt < 4; ++nt) gt2[nt] = wmma_bf16(Ag, ld_frag(W3T, kc*4 + nt, lane), gt2[nt]);
        }
        for (int nt = 0; nt < 4; ++nt)
            for (int v = 0; v < 8; ++v) { int m = v + 8*half; G2st[((nt>>1)<<9) + a_idx(m, (nt&1)*16 + col)] = f2bf(gt2[nt][v] * dsiluf(z2r[nt][v])); }
        // g_t1 = (g_t2' @ W2^T) * silu'(z1)
        v16bf Ga = ld_frag(G2st, 0, lane), Gb = ld_frag(G2st, 1, lane);
        for (int nt = 0; nt < 4; ++nt) {
            v8f a = wmma_bf16(Ga, ld_frag(W2T, nt, lane), vzero());
            a = wmma_bf16(Gb, ld_frag(W2T, 4 + nt, lane), a);
            for (int v = 0; v < 8; ++v) { int m = v + 8*half; G2st[((nt>>1)<<9) + a_idx(m, (nt&1)*16 + col)] = f2bf(a[v] * dsiluf(z1r[nt][v])); }
        }
        // g_R = g_t1 @ W1^T (cols 0..7 valid, padded cols read zeros)
        v16bf Ha = ld_frag(G2st, 0, lane), Hb = ld_frag(G2st, 1, lane);
        v8f aR = wmma_bf16(Ha, ld_frag(W1T, 0, lane), vzero());
        aR = wmma_bf16(Hb, ld_frag(W1T, 1, lane), aR);
        if (col < NBR)
            for (int v = 0; v < 8; ++v) { int m = v + 8*half; GRl[m*NBR + col] = aR[v]; }
        // radial part of forces
        if (lane < 16) {
            int m = lane;
            float ir = IVl[m], r = 1.f / ir;
            float tt = r * (1.f / RMAXF), env = 0.f, denv = 0.f;
            if (tt < 1.f) {
                float t2 = tt*tt, t4 = t2*t2, t5 = t4*tt, t6 = t5*tt, t7 = t6*tt;
                env  = 1.f - 28.f*t6 + 48.f*t7 - 21.f*t7*tt;
                denv = (-168.f*t5 + 336.f*t6 - 168.f*t7) * (1.f / RMAXF);
            }
            float gr = 0.f;
            const float Ac = 0.632455532f;
            for (int k = 0; k < NBR; ++k) {
                float a = (float)(k+1) * (PI_F / RMAXF);
                float sn = __sinf(a * r), cs = __cosf(a * r);
                float f  = Ac * sn * ir;
                float fp = Ac * (a * cs - sn * ir) * ir;
                gr += GRl[m*NBR + k] * (fp * env + f * denv);
            }
            float gvx = gr * Ul[m*3], gvy = gr * Ul[m*3+1], gvz = gr * Ul[m*3+2];
            int rc = RID[m], sd = SID[m];
            atomicAdd(&forces[rc*3+0], -gvx); atomicAdd(&forces[sd*3+0], gvx);
            atomicAdd(&forces[rc*3+1], -gvy); atomicAdd(&forces[sd*3+1], gvy);
            atomicAdd(&forces[rc*3+2], -gvz); atomicAdd(&forces[sd*3+2], gvz);
        }
    }
}

// g_h1 = scale*W_read0 + g_h2 @ W_skip1(spec)^T + g_hu1 @ W_up1^T
__global__ void k_ghprev(const float* __restrict__ gh, const float* __restrict__ ghu,
                         const float* __restrict__ Wskip1, const float* __restrict__ Wup1,
                         const float* __restrict__ Wread0, const float* __restrict__ scale,
                         const int* __restrict__ spec, float* ghn) {
    int idx = blockIdx.x * blockDim.x + threadIdx.x;
    if (idx >= NN * CC) return;
    int n = idx >> 5, c = idx & 31;
    const float* Wsk = Wskip1 + (size_t)spec[n] * CC * CC;
    float s = scale[0] * Wread0[c];
    for (int d = 0; d < CC; ++d)
        s += gh[n*CC + d] * Wsk[c*CC + d] + ghu[n*CC + d] * Wup1[c*CC + d];
    ghn[idx] = s;
}

// ---------------- host orchestration ----------------
extern "C" void kernel_launch(void* const* d_in, const int* in_sizes, int n_in,
                              void* d_out, int out_size, void* d_ws, size_t ws_size,
                              hipStream_t stream) {
    const float* pos    = (const float*)d_in[0];
    const float* attrs  = (const float*)d_in[1];
    const float* shifts = (const float*)d_in[2];
    const float* ae     = (const float*)d_in[3];
    const float* Wemb   = (const float*)d_in[4];
    const float* Wup    = (const float*)d_in[5];
    const float* Wm1    = (const float*)d_in[6];
    const float* Wm2    = (const float*)d_in[7];
    const float* Wm3    = (const float*)d_in[8];
    const float* Wlin   = (const float*)d_in[9];
    const float* Wskip  = (const float*)d_in[10];
    const float* Wprod  = (const float*)d_in[11];
    const float* Wread  = (const float*)d_in[12];
    const float* scale  = (const float*)d_in[13];
    const float* shv    = (const float*)d_in[14];
    const int*   eidx   = (const int*)d_in[15];
    const int*   batch  = (const int*)d_in[16];

    float* out  = (float*)d_out;
    float* o_te = out;                       // (G,)
    float* o_ne = out + GG;                  // (N,)
    float* o_ie = out + GG + NN;             // (G,)
    float* o_f  = out + 2*GG + NN;           // (N,3)
    float* o_ft = out + 2*GG + 4*(size_t)NN; // (N,64)

    float* ws = (float*)d_ws;
    size_t p = 0;
    float* w_u    = ws + p; p += 3*(size_t)NE;
    float* w_ivr  = ws + p; p += NE;
    float* w_Y    = ws + p; p += 9*(size_t)NE;
    float* w_R    = ws + p; p += 8*(size_t)NE;
    int*   w_spec = (int*)(ws + p); p += NN;
    float* w_h0   = ws + p; p += 32*(size_t)NN;
    float* w_h1   = ws + p; p += 32*(size_t)NN;
    float* w_h2   = ws + p; p += 32*(size_t)NN;
    float* w_hu0  = ws + p; p += 32*(size_t)NN;
    float* w_hu1  = ws + p; p += 32*(size_t)NN;
    float* w_ne0  = ws + p; p += NN;
    float* w_es   = ws + p; p += NN;
    float* w_sc   = ws + p; p += 32*(size_t)NN;
    float* w_wlp  = ws + p; p += 2*CS*CC;
    float* w_msg0 = ws + p; p += (size_t)CS*NN;
    float* w_msg1 = ws + p; p += (size_t)CS*NN;
    float* w_gh   = ws + p; p += 32*(size_t)NN;
    float* w_ghn  = ws + p; p += 32*(size_t)NN;
    float* w_ghu  = ws + p; p += 32*(size_t)NN;
    float* w_e0g  = ws + p; p += GG;
    float* w_ieg  = ws + p; p += GG + 8;
    p = (p + 7) & ~(size_t)7;                                 // 32B-align
    unsigned short* w_swz = (unsigned short*)(ws + p); p += (2*LAYER_USH + 1)/2 + 4;
    p = (p + 7) & ~(size_t)7;
    unsigned short* w_gtpw = (unsigned short*)(ws + p); p += ((size_t)NE*CS + 1)/2;
    float* w_gmsg = w_msg0;                  // msg0 dead after layer-0 update; reuse

    hipMemsetAsync(d_out, 0, (size_t)out_size * sizeof(float), stream);
    hipMemsetAsync(w_msg0, 0, 2*(size_t)CS*NN * sizeof(float), stream);  // msg0+msg1
    hipMemsetAsync(w_es,   0, (size_t)NN * sizeof(float), stream);
    hipMemsetAsync(w_ghu,  0, 32*(size_t)NN * sizeof(float), stream);
    hipMemsetAsync(w_e0g,  0, 2*GG * sizeof(float), stream);             // e0g+ieg

    k_node_init<<<(NN+255)/256, 256, 0, stream>>>(attrs, ae, Wemb, batch, w_h0, w_ne0, w_spec, w_e0g);
    k_edge_geom<<<(NE+255)/256, 256, 0, stream>>>(pos, shifts, eidx, w_u, w_ivr, w_Y, w_R);
    k_wlp<<<(2*CS*CC+255)/256, 256, 0, stream>>>(Wlin, Wprod, w_wlp);

    // ---- pre-swizzle all weight images into WMMA fragment order ----
    for (int l = 0; l < 2; ++l) {
        unsigned short* b = w_swz + (size_t)l * LAYER_USH;
        k_swz<<<(32*64+255)/256, 256, 0, stream>>>(Wm1 + l*NBR*HH, b + OFF_W1P, NBR, HH, 32, HH, 0);
        k_swz<<<(64*64+255)/256, 256, 0, stream>>>(Wm2 + l*HH*HH, b + OFF_W2F, HH, HH, HH, HH, 0);
        k_swz<<<(64*288+255)/256, 256, 0, stream>>>(Wm3 + l*HH*CS, b + OFF_W3F, HH, CS, HH, CS, 0);
        k_swz<<<(288*64+255)/256, 256, 0, stream>>>(Wm3 + l*HH*CS, b + OFF_W3T, HH, CS, CS, HH, 1);
        k_swz<<<(64*64+255)/256, 256, 0, stream>>>(Wm2 + l*HH*HH, b + OFF_W2T, HH, HH, HH, HH, 1);
        k_swz<<<(64*16+255)/256, 256, 0, stream>>>(Wm1 + l*NBR*HH, b + OFF_W1T, NBR, HH, HH, 16, 1);
        k_swz<<<(288*32+255)/256, 256, 0, stream>>>(w_wlp + l*CS*CC, b + OFF_WLP, CS, CC, CS, CC, 0);
        k_swz<<<(32*288+255)/256, 256, 0, stream>>>(w_wlp + l*CS*CC, b + OFF_WLPT, CS, CC, CC, CS, 1);
        k_swz<<<(32*32+255)/256, 256, 0, stream>>>(Wup + l*CC*CC, b + OFF_WUP, CC, CC, CC, CC, 0);
    }
    unsigned short* sw0 = w_swz;
    unsigned short* sw1 = w_swz + LAYER_USH;

    // ---- layer 0 ----
    k_hu<<<157, 256, 0, stream>>>(w_h0, sw0, w_hu0);
    k_edge_msg<<<512, 64, 0, stream>>>(sw0, w_hu0, w_Y, w_R, eidx, w_msg0);
    k_sc<<<(NN*CC+255)/256, 256, 0, stream>>>(w_h0, Wskip, w_spec, w_sc);
    k_node_update<<<157, 256, 0, stream>>>(w_msg0, sw0, w_sc, w_h1, o_ft);
    k_es<<<(NN+255)/256, 256, 0, stream>>>(w_h1, Wread, w_es);
    // ---- layer 1 ----
    k_hu<<<157, 256, 0, stream>>>(w_h1, sw1, w_hu1);
    k_edge_msg<<<512, 64, 0, stream>>>(sw1, w_hu1, w_Y, w_R, eidx, w_msg1);
    k_sc<<<(NN*CC+255)/256, 256, 0, stream>>>(w_h1, Wskip + ZZ*CC*CC, w_spec, w_sc);
    k_node_update<<<157, 256, 0, stream>>>(w_msg1, sw1, w_sc, w_h2, o_ft + 32);
    k_es<<<(NN+255)/256, 256, 0, stream>>>(w_h2, Wread + CC, w_es);

    k_readout<<<(NN+255)/256, 256, 0, stream>>>(w_es, w_ne0, scale, shv, batch, o_ne, w_ieg);
    k_final<<<1, 32, 0, stream>>>(w_e0g, w_ieg, o_te, o_ie);

    // ---- backward to forces ----
    k_ginit<<<(NN*CC+255)/256, 256, 0, stream>>>(Wread + CC, scale, w_gh);
    k_gmsg<<<157, 256, 0, stream>>>(w_gh, sw1, w_gmsg);
    k_edge_b1<<<1024, 32, 0, stream>>>(sw1, w_hu1, w_Y, w_R, w_u, w_ivr, eidx, w_gmsg, w_gtpw, o_f, w_ghu, 1);
    k_edge_b2<<<1024, 32, 0, stream>>>(sw1, w_R, w_u, w_ivr, eidx, w_gtpw, o_f);
    k_ghprev<<<(NN*CC+255)/256, 256, 0, stream>>>(w_gh, w_ghu, Wskip + ZZ*CC*CC, Wup + CC*CC,
                                                  Wread, scale, w_spec, w_ghn);
    k_gmsg<<<157, 256, 0, stream>>>(w_ghn, sw0, w_gmsg);
    k_edge_b1<<<1024, 32, 0, stream>>>(sw0, w_hu0, w_Y, w_R, w_u, w_ivr, eidx, w_gmsg, w_gtpw, o_f, w_ghu, 0);
    k_edge_b2<<<1024, 32, 0, stream>>>(sw0, w_R, w_u, w_ivr, eidx, w_gtpw, o_f);
    (void)in_sizes; (void)n_in; (void)ws_size; (void)p;
}
